// MambaBlock_19885698581120
// MI455X (gfx1250) — matrix operand order repaired and checked by reading, not compiled, not run on hardware
//
#include <hip/hip_runtime.h>
#include <hip/hip_bf16.h>

// ---------------------------------------------------------------------------
// MambaBlock for MI455X (gfx1250, wave32, WMMA, async global->LDS).
//   BATCH=8, SEQ=4096, D_MODEL=1024, D_STATE=16  -> 32768 rows of 1024.
// Pipeline:
//   prep_kernel    : w_gate -> bf16 TRANSPOSED [n][k]; C -> bf16 transposed,
//                    K zero-padded to 32  (one-time cost; removes all per-tile
//                    LDS scatter stores from the GEMM hot loop)
//   ln_proj_kernel : LayerNorm + xn(bf16) + delta=sigmoid(xn@Wd+bd), Bv=xn@Wi+bi
//   recur_kernel   : h_t = tanh((h*delta)@A + B_t), wave32-shuffle matvec
//   gemm_gate_out  : out = (hs@C) * sigmoid(xn@Wg + bg) + x
//                    128x128x32 tiles, double-buffered LDS staged with
//                    global_load_async_to_lds_b128 + s_wait_asynccnt.
// ---------------------------------------------------------------------------

#define D_MODEL 1024
#define D_STATE 16
#define NROWS   32768
#define SEQ_LEN 4096
#define NBATCH  8

typedef __attribute__((ext_vector_type(16))) __bf16 v16bf;
typedef __attribute__((ext_vector_type(8)))  float  v8f;

union V16 { v16bf v; uint4 q[2]; };

__device__ __forceinline__ unsigned short f2bf(float f) {
    union { float f; unsigned int u; } c; c.f = f;
    unsigned int r = c.u + 0x7FFFu + ((c.u >> 16) & 1u);   // round-to-nearest-even
    return (unsigned short)(r >> 16);
}

// Generic pointers to LDS on AMDGCN are {SHARED_BASE aperture, 32-bit LDS
// offset}; the async-to-LDS VDST operand wants that low-32-bit LDS address.
__device__ __forceinline__ unsigned lds_addr32(const void* p) {
    return (unsigned)(unsigned long long)p;
}

// CDNA5 async copy: 16 bytes per lane, global -> LDS, tracked by ASYNCcnt.
__device__ __forceinline__ void async_cp_b128(const void* gsrc, const void* ldst) {
    asm volatile("global_load_async_to_lds_b128 %0, %1, off"
                 :: "v"(lds_addr32(ldst)), "v"(gsrc)
                 : "memory");
}

// ---------------------------------------------------------------------------
// Kernel 0: weight conversion + transpose (one-time).
//   wgT[n*1024+k]  = bf16(wg[k*1024+n])          (1024x1024)
//   cT [n*32 +k]   = k<16 ? bf16(C[k*1024+n]) : 0 (1024x32, K padded)
// ---------------------------------------------------------------------------
__global__ __launch_bounds__(256) void prep_kernel(const float* __restrict__ wg,
                                                   const float* __restrict__ C,
                                                   unsigned short* __restrict__ wgT,
                                                   unsigned short* __restrict__ cT) {
    const int i = blockIdx.x * 256 + threadIdx.x;
    if (i < D_MODEL * D_MODEL) {
        const int n = i >> 10, k = i & 1023;
        wgT[i] = f2bf(wg[(size_t)k * D_MODEL + n]);
    }
    if (i < 32 * D_MODEL) {
        const int n = i >> 5, k = i & 31;
        cT[i] = (k < D_STATE) ? f2bf(C[(size_t)k * D_MODEL + n]) : (unsigned short)0;
    }
}

// ---------------------------------------------------------------------------
// Kernel 1: per-row LayerNorm + small projections (xn read once).
// ---------------------------------------------------------------------------
__global__ __launch_bounds__(256) void ln_proj_kernel(
    const float* __restrict__ x,
    const float* __restrict__ gamma, const float* __restrict__ beta,
    const float* __restrict__ wd, const float* __restrict__ bd,
    const float* __restrict__ wi, const float* __restrict__ bi,
    unsigned short* __restrict__ xnb,
    float* __restrict__ delta, float* __restrict__ Bv)
{
    __shared__ float red[16];
    __shared__ float r2[8][32];
    const int row = blockIdx.x;
    const int tid = threadIdx.x;

    const float4 xv = ((const float4*)(x + (size_t)row * D_MODEL))[tid];
    float s  = xv.x + xv.y + xv.z + xv.w;
    float ss = xv.x * xv.x + xv.y * xv.y + xv.z * xv.z + xv.w * xv.w;
    #pragma unroll
    for (int off = 16; off > 0; off >>= 1) {
        s  += __shfl_down(s, off);
        ss += __shfl_down(ss, off);
    }
    const int wv = tid >> 5, lane = tid & 31;
    if (lane == 0) { red[wv] = s; red[8 + wv] = ss; }
    __syncthreads();
    if (tid == 0) {
        float a = 0.f, b = 0.f;
        for (int i = 0; i < 8; ++i) { a += red[i]; b += red[8 + i]; }
        red[0] = a; red[1] = b;
    }
    __syncthreads();
    const float mu   = red[0] * (1.0f / D_MODEL);
    const float var  = red[1] * (1.0f / D_MODEL) - mu * mu;   // biased, matches jnp.var
    const float rstd = rsqrtf(var + 1e-5f);

    const int e0 = tid * 4;
    const float xin[4] = { xv.x, xv.y, xv.z, xv.w };
    float accD[16], accB[16];
    #pragma unroll
    for (int i = 0; i < 16; ++i) { accD[i] = 0.f; accB[i] = 0.f; }

    ushort4 ov;
    unsigned short* op = (unsigned short*)&ov;
    #pragma unroll
    for (int q = 0; q < 4; ++q) {
        const float v = (xin[q] - mu) * rstd * gamma[e0 + q] + beta[e0 + q];
        op[q] = f2bf(v);
        const float* wdr = wd + (size_t)(e0 + q) * D_STATE;
        const float* wir = wi + (size_t)(e0 + q) * D_STATE;
        #pragma unroll
        for (int s2 = 0; s2 < 16; ++s2) {
            accD[s2] = fmaf(v, wdr[s2], accD[s2]);
            accB[s2] = fmaf(v, wir[s2], accB[s2]);
        }
    }
    ((ushort4*)(xnb + (size_t)row * D_MODEL))[tid] = ov;

    #pragma unroll
    for (int s2 = 0; s2 < 16; ++s2) {
        #pragma unroll
        for (int off = 16; off > 0; off >>= 1) {
            accD[s2] += __shfl_down(accD[s2], off);
            accB[s2] += __shfl_down(accB[s2], off);
        }
    }
    if (lane == 0) {
        #pragma unroll
        for (int s2 = 0; s2 < 16; ++s2) { r2[wv][s2] = accD[s2]; r2[wv][16 + s2] = accB[s2]; }
    }
    __syncthreads();
    if (tid < 32) {
        float a = 0.f;
        for (int w = 0; w < 8; ++w) a += r2[w][tid];
        if (tid < 16) delta[(size_t)row * 16 + tid] = 1.0f / (1.0f + __expf(-(a + bd[tid])));
        else          Bv[(size_t)row * 16 + (tid - 16)] = a + bi[tid - 16];
    }
}

// ---------------------------------------------------------------------------
// Kernel 2: sequential recurrence. 128 threads = 8 batches x 16 lanes.
// ---------------------------------------------------------------------------
__global__ __launch_bounds__(128) void recur_kernel(const float* __restrict__ A,
                                                    const float* __restrict__ delta,
                                                    const float* __restrict__ Bv,
                                                    unsigned short* __restrict__ hs)
{
    const int t0    = threadIdx.x;
    const int batch = t0 >> 4;
    const int j     = t0 & 15;
    const int base  = t0 & 16;           // 0 or 16 within the wave32

    float Acol[16];
    #pragma unroll
    for (int i = 0; i < 16; ++i) Acol[i] = A[i * 16 + j];

    float h = 0.f;
    const size_t rb = (size_t)batch * SEQ_LEN * 16;
    for (int t = 0; t < SEQ_LEN; ++t) {
        const size_t r = rb + (size_t)t * 16;
        const float d  = delta[r + j];
        const float bt = Bv[r + j];
        const float hd = h * d;
        float acc = bt;
        #pragma unroll
        for (int i = 0; i < 16; ++i)
            acc = fmaf(__shfl(hd, base + i, 32), Acol[i], acc);
        h = tanhf(acc);
        const size_t hr = ((size_t)batch * SEQ_LEN + t) * 32;
        hs[hr + j]      = f2bf(h);
        hs[hr + 16 + j] = 0;
    }
}

// ---------------------------------------------------------------------------
// Kernel 3: fused WMMA GEMM with double-buffered async LDS staging.
//   acc  = xn(bf16) @ Wg        (A: xn rows; B: wgT rows -> both contiguous)
//   acc2 = hs(bf16,Kpad32) @ C  (single K-tile, same operand paths)
//   out  = acc2 * sigmoid(acc + bg) + x
// Per k-step each thread issues 4x global_load_async_to_lds_b128 for the NEXT
// tile, then s_wait_asynccnt 4 guarantees the CURRENT tile is resident
// (async loads complete in order).
// ---------------------------------------------------------------------------
__global__ __launch_bounds__(256) void gemm_gate_out_kernel(
    const unsigned short* __restrict__ xn,
    const unsigned short* __restrict__ wgT,
    const unsigned short* __restrict__ hsb,
    const unsigned short* __restrict__ cT,
    const float* __restrict__ bgate,
    const float* __restrict__ x,
    float* __restrict__ out)
{
    __shared__ __align__(16) unsigned short aT[2][128 * 32];
    __shared__ __align__(16) unsigned short bT[2][128 * 32];

    const int tid  = threadIdx.x;
    const int bm   = (int)(blockIdx.x >> 3) * 128;
    const int bn   = (int)(blockIdx.x & 7) * 128;
    const int lane = tid & 31;
    const int wave = tid >> 5;
    const int wM   = (wave & 1) * 64;
    const int wN   = (wave >> 1) * 32;
    const int hsel = lane >> 4;
    const int ln   = lane & 15;

    const int srow  = tid >> 1;          // 0..127 : tile row (m for A, n for B)
    const int shalf = (tid & 1) * 16;    // 0 or 16 halfs within the 32-wide row

    const v8f zero = {};
    v8f acc[4][2];
    #pragma unroll
    for (int i = 0; i < 4; ++i)
        #pragma unroll
        for (int j = 0; j < 2; ++j) acc[i][j] = zero;

    // ---- tile staging: 4 async b128 per thread ------------------------
    auto stage = [&](int kt, int buf) {
        const unsigned short* ga = xn  + (size_t)(bm + srow) * D_MODEL + kt * 32 + shalf;
        const unsigned short* gb = wgT + (size_t)(bn + srow) * D_MODEL + kt * 32 + shalf;
        unsigned short* la = &aT[buf][srow * 32 + shalf];
        unsigned short* lb = &bT[buf][srow * 32 + shalf];
        async_cp_b128(ga,     la);
        async_cp_b128(ga + 8, la + 8);
        async_cp_b128(gb,     lb);
        async_cp_b128(gb + 8, lb + 8);
    };

    stage(0, 0);
    for (int kt = 0; kt < 32; ++kt) {
        const int cur = kt & 1;
        if (kt < 31) {
            stage(kt + 1, cur ^ 1);
            asm volatile("s_wait_asynccnt 4" ::: "memory");
        } else {
            asm volatile("s_wait_asynccnt 0" ::: "memory");
        }
        __syncthreads();

        V16 b0, b1;
        b0.q[0] = *(const uint4*)(&bT[cur][(wN + ln) * 32 + hsel * 16]);
        b0.q[1] = *(const uint4*)(&bT[cur][(wN + ln) * 32 + hsel * 16 + 8]);
        b1.q[0] = *(const uint4*)(&bT[cur][(wN + 16 + ln) * 32 + hsel * 16]);
        b1.q[1] = *(const uint4*)(&bT[cur][(wN + 16 + ln) * 32 + hsel * 16 + 8]);
        #pragma unroll
        for (int tm = 0; tm < 4; ++tm) {
            V16 a;
            const int m = wM + tm * 16 + ln;
            a.q[0] = *(const uint4*)(&aT[cur][m * 32 + hsel * 16]);
            a.q[1] = *(const uint4*)(&aT[cur][m * 32 + hsel * 16 + 8]);
            acc[tm][0] = __builtin_amdgcn_wmma_f32_16x16x32_bf16(
                false, a.v, false, b0.v, (short)0, acc[tm][0], false, false);
            acc[tm][1] = __builtin_amdgcn_wmma_f32_16x16x32_bf16(
                false, a.v, false, b1.v, (short)0, acc[tm][1], false, false);
        }
        __syncthreads();   // all waves done with buf `cur` before it is restaged
    }

    // -------- epilogue GEMM: hs @ C (single K-tile, K padded to 32) -----
    {
        const unsigned short* gh = hsb + (size_t)(bm + srow) * 32 + shalf;
        const unsigned short* gc = cT  + (size_t)(bn + srow) * 32 + shalf;
        unsigned short* la = &aT[0][srow * 32 + shalf];
        unsigned short* lb = &bT[0][srow * 32 + shalf];
        async_cp_b128(gh,     la);
        async_cp_b128(gh + 8, la + 8);
        async_cp_b128(gc,     lb);
        async_cp_b128(gc + 8, lb + 8);
        asm volatile("s_wait_asynccnt 0" ::: "memory");
    }
    __syncthreads();

    v8f acc2[4][2];
    #pragma unroll
    for (int i = 0; i < 4; ++i)
        #pragma unroll
        for (int j = 0; j < 2; ++j) acc2[i][j] = zero;
    {
        V16 b0, b1;
        b0.q[0] = *(const uint4*)(&bT[0][(wN + ln) * 32 + hsel * 16]);
        b0.q[1] = *(const uint4*)(&bT[0][(wN + ln) * 32 + hsel * 16 + 8]);
        b1.q[0] = *(const uint4*)(&bT[0][(wN + 16 + ln) * 32 + hsel * 16]);
        b1.q[1] = *(const uint4*)(&bT[0][(wN + 16 + ln) * 32 + hsel * 16 + 8]);
        #pragma unroll
        for (int tm = 0; tm < 4; ++tm) {
            V16 a;
            const int m = wM + tm * 16 + ln;
            a.q[0] = *(const uint4*)(&aT[0][m * 32 + hsel * 16]);
            a.q[1] = *(const uint4*)(&aT[0][m * 32 + hsel * 16 + 8]);
            acc2[tm][0] = __builtin_amdgcn_wmma_f32_16x16x32_bf16(
                false, a.v, false, b0.v, (short)0, acc2[tm][0], false, false);
            acc2[tm][1] = __builtin_amdgcn_wmma_f32_16x16x32_bf16(
                false, a.v, false, b1.v, (short)0, acc2[tm][1], false, false);
        }
    }

    // -------- write: out = acc2 * sigmoid(acc + bg) + x -----------------
    // C/D layout: VGPR v -> M = v + 8*hsel ; N = lane&15.
    #pragma unroll
    for (int tm = 0; tm < 4; ++tm)
        #pragma unroll
        for (int tn = 0; tn < 2; ++tn)
            #pragma unroll
            for (int v = 0; v < 8; ++v) {
                const int gm = bm + wM + tm * 16 + v + 8 * hsel;
                const int ge = bn + wN + tn * 16 + ln;
                const float g  = acc[tm][tn][v] + bgate[ge];
                const float sg = 1.0f / (1.0f + __expf(-g));
                const size_t o = (size_t)gm * D_MODEL + ge;
                out[o] = acc2[tm][tn][v] * sg + x[o];
            }
}

// ---------------------------------------------------------------------------
extern "C" void kernel_launch(void* const* d_in, const int* in_sizes, int n_in,
                              void* d_out, int out_size, void* d_ws, size_t ws_size,
                              hipStream_t stream)
{
    const float* x   = (const float*)d_in[0];
    const float* gam = (const float*)d_in[1];
    const float* bet = (const float*)d_in[2];
    const float* A   = (const float*)d_in[3];
    const float* C   = (const float*)d_in[4];
    const float* wd  = (const float*)d_in[5];
    const float* bd  = (const float*)d_in[6];
    const float* wi  = (const float*)d_in[7];
    const float* bi  = (const float*)d_in[8];
    const float* wg  = (const float*)d_in[9];
    const float* bg  = (const float*)d_in[10];
    float* out = (float*)d_out;

    // workspace layout (bytes):
    //   xn  bf16      : 32768*1024*2 = 67,108,864
    //   wgT bf16 [n][k]:  1024*1024*2 =  2,097,152
    //   cT  bf16 [n][32]:   1024*32*2 =     65,536
    //   delta f32     : 32768*16*4   =  2,097,152
    //   Bv    f32     : 32768*16*4   =  2,097,152
    //   hs  bf16      : 32768*32*2   =  2,097,152   (total ~75.6 MB)
    char* ws = (char*)d_ws;
    unsigned short* xnb = (unsigned short*)(ws);
    unsigned short* wgT = (unsigned short*)(ws + 67108864ull);
    unsigned short* cT  = (unsigned short*)(ws + 67108864ull + 2097152ull);
    float* delta        = (float*)(ws + 67108864ull + 2097152ull + 65536ull);
    float* Bv           = (float*)(ws + 67108864ull + 2097152ull + 65536ull + 2097152ull);
    unsigned short* hsb = (unsigned short*)(ws + 67108864ull + 2097152ull + 65536ull + 2097152ull + 2097152ull);

    prep_kernel<<<dim3(4096), dim3(256), 0, stream>>>(wg, C, wgT, cT);
    ln_proj_kernel<<<dim3(NROWS), dim3(256), 0, stream>>>(x, gam, bet, wd, bd, wi, bi,
                                                          xnb, delta, Bv);
    recur_kernel<<<dim3(1), dim3(128), 0, stream>>>(A, delta, Bv, hsb);
    gemm_gate_out_kernel<<<dim3(2048), dim3(256), 0, stream>>>(xnb, wgT, hsb, cT, bg, x, out);

    (void)in_sizes; (void)n_in; (void)out_size; (void)ws_size;
}
